// HWnet_plus_21251498180926
// MI455X (gfx1250) — matrix operand order repaired and checked by reading, not compiled, not run on hardware
//
#include <hip/hip_runtime.h>
#include <hip/hip_bf16.h>
#include <stdint.h>

typedef __attribute__((ext_vector_type(16))) __bf16 v16bf;
typedef __attribute__((ext_vector_type(8)))  float  v8f;

#define NUM_BINS   2048
#define EMBED      128
#define NBLOCKS    128          // 2048 bins / 16-bin slab granularity
#define SPLIT      8            // workgroups per slab
#define NWAVES     8            // waves per workgroup (256 threads, wave32)
#define TAKECARE   4.0f
#define LOV        (-4.0f)
#define BIN_W      0.00390625f  // 8/2048, exact in f32
#define INV_BIN_W  256.0f

__device__ __forceinline__ int bin_index(float xv) {
    int idx = (int)floorf((xv - LOV) * INV_BIN_W);
    idx = idx < 0 ? 0 : (idx > NUM_BINS - 1 ? NUM_BINS - 1 : idx);
    return idx;
}
__device__ __forceinline__ int clamp_idx(int idx) {
    return idx < 4 ? 4 : (idx > NUM_BINS - 5 ? NUM_BINS - 5 : idx);
}

// ---------------- phase 1: histogram samples by 16-aligned slab ----------------
__global__ void k_hist(const float* __restrict__ x, int* __restrict__ hist,
                       int* __restrict__ block_of, int batch) {
    int i = blockIdx.x * blockDim.x + threadIdx.x;
    if (i >= batch) return;
    int idxc = clamp_idx(bin_index(x[i]));
    int blk  = (idxc - 4) >> 4;          // 0..127
    block_of[i] = blk;
    atomicAdd(&hist[blk], 1);
}

// ---------------- phase 2: exclusive scan over 128 counters --------------------
__global__ void k_scan(const int* __restrict__ hist, int* __restrict__ offsets) {
    if (threadIdx.x == 0) {
        int acc = 0;
        for (int b = 0; b < NBLOCKS; ++b) { offsets[b] = acc; acc += hist[b]; }
        offsets[NBLOCKS] = acc;
    }
}

// ---------------- phase 3: scatter sample ids into slab order ------------------
__global__ void k_scatter(const int* __restrict__ block_of, const int* __restrict__ offsets,
                          int* __restrict__ counter, int* __restrict__ sorted, int batch) {
    int i = blockIdx.x * blockDim.x + threadIdx.x;
    if (i >= batch) return;
    int blk = block_of[i];
    int pos = offsets[blk] + atomicAdd(&counter[blk], 1);
    sorted[pos] = i;
}

// ---------------- phase 4: WMMA — 16-sample groups vs shared 32x128 slab -------
__global__ void __launch_bounds__(256)
k_wmma(const float* __restrict__ x, const float* __restrict__ vtab,
       const int* __restrict__ hist, const int* __restrict__ offsets,
       const int* __restrict__ sorted, float* __restrict__ out) {
    __shared__ __align__(16) __bf16 Bt[EMBED][32];          // table slab, transposed (8 KB)
    __shared__ __align__(16) __bf16 Arow[NWAVES][16][32];   // per-wave score tiles (8 KB)
    __shared__ int sids[NWAVES][16];

    const int wg   = blockIdx.x;
    const int blk  = wg / SPLIT;
    const int sub  = wg % SPLIT;
    const int tid  = threadIdx.x;
    const int wave = tid >> 5;
    const int lane = tid & 31;
    const int bstart = blk << 4;

    // Stage V[bstart .. bstart+31][0..127] -> bf16, transposed. Coalesced 512B reads.
    for (int t = tid; t < 32 * EMBED; t += 256) {
        int e = t & (EMBED - 1);
        int k = t >> 7;                 // 0..31
        Bt[e][k] = (__bf16)vtab[(bstart + k) * EMBED + e];
    }
    __syncthreads();

    const int n   = hist[blk];
    const int off = offsets[blk];
    const int totalGroups = (n + 15) >> 4;
    const int gw = sub * NWAVES + wave;         // global wave slot within this slab

    union UF { uint32_t u[8]; v16bf v; };
    const int ncol = lane & 15;
    const int half = lane >> 4;

    // ---- B fragments are loop-invariant: hoist all 8 N-tiles into registers.
    // 32x16 bf16 B layout: lanes 0-15 hold K0-15, lanes 16-31 hold K16-31.
    UF ub[EMBED / 16];
    #pragma unroll
    for (int nt = 0; nt < EMBED / 16; ++nt) {
        const uint4* bp = (const uint4*)&Bt[nt * 16 + ncol][0];
        uint4 b0 = bp[half * 2];
        uint4 b1 = bp[half * 2 + 1];
        ub[nt].u[0]=b0.x; ub[nt].u[1]=b0.y; ub[nt].u[2]=b0.z; ub[nt].u[3]=b0.w;
        ub[nt].u[4]=b1.x; ub[nt].u[5]=b1.y; ub[nt].u[6]=b1.z; ub[nt].u[7]=b1.w;
    }

    for (int g = gw; g < totalGroups; g += SPLIT * NWAVES) {
        // ---- per-sample softmax scores -> A tile rows (lanes 0-15, one sample each).
        // Tail padding: duplicate the slab's last sample -> identical values are
        // rewritten to the same output row (benign, deterministic). No invalid slots.
        if (lane < 16) {
            int slot = g * 16 + lane;
            if (slot >= n) slot = n - 1;
            int sid  = sorted[off + slot];
            sids[wave][lane] = sid;
            uint4 z = make_uint4(0u, 0u, 0u, 0u);
            uint4* rp = (uint4*)&Arow[wave][lane][0];
            rp[0] = z; rp[1] = z; rp[2] = z; rp[3] = z;   // zero 32 bf16

            float xv  = x[sid];
            int  idx  = bin_index(xv);
            int  idxc = clamp_idx(idx);
            float ev   = LOV + ((float)idx + 0.5f) * BIN_W;        // bin center
            float base = (xv - ev) * INV_BIN_W - (float)(idxc - idx);
            float l[9]; float mx = -1e30f;
            #pragma unroll
            for (int j = 0; j < 9; ++j) {
                float d = base - (float)(j - 4);
                l[j] = -TAKECARE * d * d;
                mx = fmaxf(mx, l[j]);
            }
            float s[9]; float sum = 0.0f;
            #pragma unroll
            for (int j = 0; j < 9; ++j) { s[j] = __expf(l[j] - mx); sum += s[j]; }
            float inv = 1.0f / sum;
            int col0 = (idxc - 4) - bstart;                        // 0..15
            #pragma unroll
            for (int j = 0; j < 9; ++j)
                Arow[wave][lane][col0 + j] = (__bf16)(s[j] * inv);
        }
        // wave-local LDS handoff (writers: lanes 0-15; readers: all 32 lanes)
        asm volatile("s_wait_dscnt 0" ::: "memory");

        // Per-lane output row bases (ISA C/D layout: lanes 0-15 hold rows r,
        // lanes 16-31 hold rows r+8; VGPR r = row). All rows valid -> no guards.
        int m0 = half ? 8 : 0;
        int rowbase[8];
        #pragma unroll
        for (int r = 0; r < 8; ++r)
            rowbase[r] = sids[wave][m0 + r] * EMBED + ncol;

        // ---- A fragment (16x32 bf16): lanes 0-15: K0-15; lanes 16-31: K16-31
        const uint4* ap = (const uint4*)&Arow[wave][ncol][0];  // 4x16B units: K0-7,8-15,16-23,24-31
        uint4 a0 = ap[half * 2];
        uint4 a1 = ap[half * 2 + 1];
        UF ua;
        ua.u[0]=a0.x; ua.u[1]=a0.y; ua.u[2]=a0.z; ua.u[3]=a0.w;
        ua.u[4]=a1.x; ua.u[5]=a1.y; ua.u[6]=a1.z; ua.u[7]=a1.w;

        #pragma unroll
        for (int nt = 0; nt < EMBED / 16; ++nt) {
            v8f acc = {};
            acc = __builtin_amdgcn_wmma_f32_16x16x32_bf16(
                      /*neg_a=*/false, ua.v, /*neg_b=*/false, ub[nt].v,
                      /*c_mod=*/(short)0, acc, /*reuse_a=*/false, /*reuse_b=*/false);

            // Unconditional scatter: 16 lanes write consecutive floats of one
            // output row (64B bursts), 8 rows per half-wave.
            #pragma unroll
            for (int r = 0; r < 8; ++r)
                out[rowbase[r] + nt * 16] = acc[r];
        }
    }
}

extern "C" void kernel_launch(void* const* d_in, const int* in_sizes, int n_in,
                              void* d_out, int out_size, void* d_ws, size_t ws_size,
                              hipStream_t stream) {
    const float* x    = (const float*)d_in[0];   // (BATCH,1) f32
    const float* vtab = (const float*)d_in[4];   // (2048,128) f32
    float* out = (float*)d_out;                  // (BATCH,128) f32
    const int batch = in_sizes[0];

    // workspace layout (ints): hist[128] | counter[128] | offsets[129+pad] | block_of[B] | sorted[B]
    int* hist     = (int*)d_ws;
    int* counter  = hist + NBLOCKS;
    int* offsets  = counter + NBLOCKS;
    int* block_of = offsets + NBLOCKS + 4;
    int* sorted   = block_of + batch;

    hipMemsetAsync(d_ws, 0, 2 * NBLOCKS * sizeof(int), stream);   // zero hist + counter

    const int thr = 256;
    const int grid = (batch + thr - 1) / thr;
    k_hist   <<<grid, thr, 0, stream>>>(x, hist, block_of, batch);
    k_scan   <<<1, 32, 0, stream>>>(hist, offsets);
    k_scatter<<<grid, thr, 0, stream>>>(block_of, offsets, counter, sorted, batch);
    k_wmma   <<<NBLOCKS * SPLIT, 256, 0, stream>>>(x, vtab, hist, offsets, sorted, out);
}
